// TRANSSTNBLOCK_57466662420731
// MI455X (gfx1250) — matrix operand order repaired and checked by reading, not compiled, not run on hardware
//
#include <hip/hip_runtime.h>

typedef __attribute__((ext_vector_type(16))) _Float16 v16h;
typedef __attribute__((ext_vector_type(8)))  _Float16 v8h;
typedef __attribute__((ext_vector_type(8)))  float    v8f;

#define NB   4
#define CCH  128
#define HH   160
#define WW   160
#define HW   (HH*WW)        // 25600
#define NPIX (NB*HW)        // 102400
#define KTOT 1152           // 9*128
#define MT   32             // pixels per block (two 16-row WMMA M tiles)

// ---------------- stats zero ----------------
__global__ void zero_stats_kernel(float* gstats) {
    int t = threadIdx.x;
    if (t < NB*32*2) gstats[t] = 0.0f;
}

// ---------------- NCHW f32 -> NHWC f16 transpose ----------------
__global__ __launch_bounds__(256) void transpose_kernel(const float* __restrict__ x,
                                                        _Float16* __restrict__ xt) {
    __shared__ float tile[16][17];
    int bx = blockIdx.x;           // h*10 + wtile
    int h  = bx / 10;
    int wt = bx % 10;
    int ct = blockIdx.y;           // 0..7
    int n  = blockIdx.z;           // 0..3
    int t  = threadIdx.x;
    int r  = t >> 4;               // 0..15
    int cc = t & 15;               // 0..15
    tile[r][cc] = x[(((long)n*CCH + ct*16 + r)*HH + h)*WW + wt*16 + cc];
    __syncthreads();
    xt[(((long)n*HH + h)*WW + wt*16 + r)*CCH + ct*16 + cc] = (_Float16)tile[cc][r];
}

// ---------------- offset convs -> per-pixel sample coords ----------------
__global__ __launch_bounds__(256) void offsets_kernel(const _Float16* __restrict__ xt,
                                                      const float* __restrict__ w_tm,
                                                      const float* __restrict__ b_tm,
                                                      const float* __restrict__ w_tr,
                                                      const float* __restrict__ b_tr,
                                                      float2* __restrict__ pyx) {
    int p = blockIdx.x * 256 + threadIdx.x;
    if (p >= NPIX) return;
    int n = p / HW;
    int q = p - n * HW;
    int h = q / WW;
    int w = q - h * WW;
    float t00 = b_tm[0], t01 = b_tm[1], t10 = b_tm[2], t11 = b_tm[3];
    float r0  = b_tr[0], r1  = b_tr[1];
    for (int kk = 0; kk < 9; kk++) {
        int yy = h + kk/3 - 1;
        int xx = w + kk%3 - 1;
        if (yy < 0 || yy >= HH || xx < 0 || xx >= WW) continue;
        const _Float16* xr = xt + (((long)n*HH + yy)*WW + xx) * CCH;
        for (int c = 0; c < CCH; c++) {
            float xv = (float)xr[c];
            t00 += xv * w_tm[(0*CCH + c)*9 + kk];
            t01 += xv * w_tm[(1*CCH + c)*9 + kk];
            t10 += xv * w_tm[(2*CCH + c)*9 + kk];
            t11 += xv * w_tm[(3*CCH + c)*9 + kk];
            r0  += xv * w_tr[(0*CCH + c)*9 + kk];
            r1  += xv * w_tr[(1*CCH + c)*9 + kk];
        }
    }
    for (int kk = 0; kk < 9; kk++) {
        float R0 = (float)(kk/3) - 1.0f;
        float R1 = (float)(kk%3) - 1.0f;
        float py = (float)h + t00*R0 + t01*R1 + r0;
        float px = (float)w + t10*R0 + t11*R1 + r1;
        pyx[p*9 + kk] = make_float2(py, px);
    }
}

// ---------------- pack w_dcn into f16 B-fragment order ----------------
__global__ __launch_bounds__(256) void wpack_kernel(const float* __restrict__ w_dcn,
                                                    _Float16* __restrict__ wpack) {
    int idx = blockIdx.x * 256 + threadIdx.x;
    if (idx >= 9*8*4*32*16) return;
    int e  = idx & 15;
    int l  = (idx >> 4) & 31;
    int kb = (idx >> 9) & 3;
    int nt = (idx >> 11) & 7;
    int k  = idx >> 14;
    int c  = kb*32 + (l >> 4)*16 + e;   // K index = input channel
    int o  = nt*16 + (l & 15);          // N index = output channel
    wpack[idx] = (_Float16)w_dcn[(o*CCH + c)*9 + k];
}

// ---------------- main deformable-conv GEMM with WMMA ----------------
__global__ __launch_bounds__(256) void deform_wmma_kernel(const _Float16* __restrict__ xt,
                                                          const float2* __restrict__ pyx,
                                                          const _Float16* __restrict__ wpack,
                                                          float* __restrict__ outp,
                                                          float* __restrict__ gstats) {
    __shared__ __align__(16) _Float16 Alds[MT * KTOT];   // 72 KB
    __shared__ float sred[64];
    int tid  = threadIdx.x;
    int tile = blockIdx.x;
    int p0   = tile * MT;          // first pixel of tile (25600 % 32 == 0: never crosses image)
    int n    = p0 / HW;
    if (tid < 64) sred[tid] = 0.0f;

    // ---- gather phase: 32 pixels x 9 taps x 128 ch, 2 pixels x 8 ch per thread ----
    int ci = (tid & 15) * 8;       // channel chunk base
    long nbase = (long)n * HW * CCH;
    #pragma unroll
    for (int sub = 0; sub < 2; sub++) {
        int m = (tid >> 4) + sub*16;   // pixel within tile 0..31
        int p = p0 + m;
        for (int k = 0; k < 9; k++) {
            float2 pp = pyx[p*9 + k];
            float py = pp.x, px = pp.y;
            float yf = floorf(py), xf = floorf(px);
            float wy = py - yf,    wx = px - xf;
            int y0 = (int)yf, x0 = (int)xf;
            int y1 = y0 + 1,  x1 = x0 + 1;
            float vy0 = (y0 >= 0 && y0 < HH) ? 1.0f : 0.0f;
            float vy1 = (y1 >= 0 && y1 < HH) ? 1.0f : 0.0f;
            float vx0 = (x0 >= 0 && x0 < WW) ? 1.0f : 0.0f;
            float vx1 = (x1 >= 0 && x1 < WW) ? 1.0f : 0.0f;
            int yc0 = min(max(y0, 0), HH-1), yc1 = min(max(y1, 0), HH-1);
            int xc0 = min(max(x0, 0), WW-1), xc1 = min(max(x1, 0), WW-1);
            float c00 = (1.0f-wy)*(1.0f-wx)*vy0*vx0;
            float c01 = (1.0f-wy)*wx       *vy0*vx1;
            float c10 = wy*(1.0f-wx)       *vy1*vx0;
            float c11 = wy*wx              *vy1*vx1;
            const _Float16* b00 = xt + nbase + (yc0*WW + xc0)*CCH + ci;
            const _Float16* b01 = xt + nbase + (yc0*WW + xc1)*CCH + ci;
            const _Float16* b10 = xt + nbase + (yc1*WW + xc0)*CCH + ci;
            const _Float16* b11 = xt + nbase + (yc1*WW + xc1)*CCH + ci;
            v8h q00 = *(const v8h*)b00;
            v8h q01 = *(const v8h*)b01;
            v8h q10 = *(const v8h*)b10;
            v8h q11 = *(const v8h*)b11;
            v8h r;
            #pragma unroll
            for (int j = 0; j < 8; j++) {
                float s = c00*(float)q00[j] + c01*(float)q01[j]
                        + c10*(float)q10[j] + c11*(float)q11[j];
                r[j] = (_Float16)s;
            }
            *(v8h*)(Alds + m*KTOT + k*CCH + ci) = r;
        }
    }
    __syncthreads();

    // ---- WMMA phase: 8 waves; wave wv owns out-channel tile wv*16..+15,
    //      two M tiles (pixels 0..15 and 16..31) -> two independent acc chains,
    //      each B fragment reused for both M tiles ----
    int wv = tid >> 5;
    int l  = tid & 31;
    int lm = l & 15;
    int hi = l >> 4;
    v8f acc0 = {};
    v8f acc1 = {};
    for (int k = 0; k < 9; k++) {
        #pragma unroll
        for (int kb = 0; kb < 4; kb++) {
            // B fragment: pre-packed, one contiguous 32-byte run per lane
            v16h B = *(const v16h*)(wpack + ((((k*8 + wv)*4 + kb)*32 + l) << 4));
            // A fragments: 16x32 f16; lane lm = row M; two aligned 8-half runs each
            const _Float16* ab0 = Alds + lm*KTOT + k*CCH + kb*32;
            const _Float16* ab1 = ab0 + 16*KTOT;
            v8h a0lo = *(const v8h*)(ab0 + hi*16);
            v8h a0hi = *(const v8h*)(ab0 + hi*16 + 8);
            v8h a1lo = *(const v8h*)(ab1 + hi*16);
            v8h a1hi = *(const v8h*)(ab1 + hi*16 + 8);
            v16h A0, A1;
            #pragma unroll
            for (int j = 0; j < 8; j++) {
                A0[j] = a0lo[j]; A0[j+8] = a0hi[j];
                A1[j] = a1lo[j]; A1[j+8] = a1hi[j];
            }
            acc0 = __builtin_amdgcn_wmma_f32_16x16x32_f16(
                       false, A0, false, B, (short)0, acc0, false, false);
            acc1 = __builtin_amdgcn_wmma_f32_16x16x32_f16(
                       false, A1, false, B, (short)0, acc1, false, false);
        }
    }

    // ---- write pre-GN output (NCHW) + group stats ----
    int o = wv*16 + lm;                      // out channel (D col = lane)
    long obase = ((long)n*CCH + o) * HW;
    int  q0 = p0 - n*HW;
    float s1 = 0.0f, s2 = 0.0f;
    #pragma unroll
    for (int r = 0; r < 8; r++) {
        int M = r + hi*8;
        float v0 = acc0[r];
        float v1 = acc1[r];
        outp[obase + q0 + M]      = v0;
        outp[obase + q0 + 16 + M] = v1;
        s1 += v0 + v1;
        s2 += v0*v0 + v1*v1;
    }
    int g = o >> 2;                          // 4 channels per group
    atomicAdd(&sred[g*2 + 0], s1);
    atomicAdd(&sred[g*2 + 1], s2);
    __syncthreads();
    if (tid < 64) atomicAdd(&gstats[n*64 + tid], sred[tid]);
}

// ---------------- GroupNorm + residual + ReLU (in place on d_out) ----------------
__global__ __launch_bounds__(256) void finalize_kernel(const float* __restrict__ x,
                                                       const float* __restrict__ gamma,
                                                       const float* __restrict__ beta,
                                                       const float* __restrict__ gstats,
                                                       float* __restrict__ outp) {
    long idx = (long)blockIdx.x * 256 + threadIdx.x;
    if (idx >= (long)NB*CCH*HW) return;
    int n   = (int)(idx / ((long)CCH*HW));
    long rem = idx - (long)n*CCH*HW;
    int o   = (int)(rem / HW);
    int g   = o >> 2;
    float s1 = gstats[n*64 + g*2 + 0];
    float s2 = gstats[n*64 + g*2 + 1];
    const float cnt = (float)(4*HW);
    float mean = s1 / cnt;
    float var  = s2 / cnt - mean*mean;
    float inv  = rsqrtf(var + 1e-5f);
    float v = (outp[idx] - mean) * inv * gamma[o] + beta[o] + x[idx];
    outp[idx] = fmaxf(v, 0.0f);
}

extern "C" void kernel_launch(void* const* d_in, const int* in_sizes, int n_in,
                              void* d_out, int out_size, void* d_ws, size_t ws_size,
                              hipStream_t stream) {
    const float* x       = (const float*)d_in[0];
    const float* w_tm    = (const float*)d_in[1];
    const float* b_tm    = (const float*)d_in[2];
    const float* w_tr    = (const float*)d_in[3];
    const float* b_tr    = (const float*)d_in[4];
    const float* w_dcn   = (const float*)d_in[5];
    const float* gamma   = (const float*)d_in[6];
    const float* beta    = (const float*)d_in[7];
    float* outp = (float*)d_out;

    char* wsb = (char*)d_ws;
    _Float16* xt    = (_Float16*)wsb;                              // 26,214,400 B
    float2*   pyx   = (float2*)(wsb + 26214400);                   //  7,372,800 B
    _Float16* wpack = (_Float16*)(wsb + 26214400 + 7372800);       //    294,912 B
    float*    gstats= (float*)(wsb + 26214400 + 7372800 + 294912); //      1,024 B

    zero_stats_kernel<<<1, 256, 0, stream>>>(gstats);
    transpose_kernel<<<dim3(1600, 8, 4), 256, 0, stream>>>(x, xt);
    offsets_kernel<<<(NPIX + 255)/256, 256, 0, stream>>>(xt, w_tm, b_tm, w_tr, b_tr, pyx);
    wpack_kernel<<<(9*8*4*32*16 + 255)/256, 256, 0, stream>>>(w_dcn, wpack);
    deform_wmma_kernel<<<NPIX/MT, 256, 0, stream>>>(xt, pyx, wpack, outp, gstats);
    finalize_kernel<<<((long)NB*CCH*HW + 255)/256, 256, 0, stream>>>(x, gamma, beta, gstats, outp);
}
